// Net_3212635537426
// MI455X (gfx1250) — matrix operand order repaired and checked by reading, not compiled, not run on hardware
//
#include <hip/hip_runtime.h>
#include <hip/hip_bf16.h>
#include <math.h>

typedef __attribute__((ext_vector_type(16))) _Float16 v16h;
typedef __attribute__((ext_vector_type(8)))  _Float16 v8h;
typedef __attribute__((ext_vector_type(8)))  float    v8f;
typedef _Float16 h16;

#define WAVES 2   // waves per block; each wave owns a 16-image tile

// ---------------------------------------------------------------------------
// branch-free tanh: 1 - 2/(e^{2x}+1); saturates to +/-1 at large |x|
// ---------------------------------------------------------------------------
__device__ __forceinline__ float tanh_fast(float v) {
#if __has_builtin(__builtin_amdgcn_exp2f)
  float e = __builtin_amdgcn_exp2f(v * 2.8853900817779268f);
#else
  float e = exp2f(v * 2.8853900817779268f);
#endif
#if __has_builtin(__builtin_amdgcn_rcpf)
  return 1.f - 2.f * __builtin_amdgcn_rcpf(e + 1.f);
#else
  return 1.f - 2.f / (e + 1.f);
#endif
}

__device__ __forceinline__ v8f wmma_f16(v16h a, v16h b, v8f c) {
  return __builtin_amdgcn_wmma_f32_16x16x32_f16(false, a, false, b,
                                                (short)0, c, false, false);
}

// B fragment from TRANSPOSED weights Wt[n*Kpad + k]: per lane two 8-half runs
// at (n0+(l&15))*Kpad + k0 + 8*(l>=16) and +16  -> two b128 loads.
__device__ __forceinline__ v16h load_bfrag(const h16* __restrict__ Wt, int Kpad,
                                           int n0, int k0, int ncol, int hs) {
  const h16* p = Wt + (n0 + ncol) * Kpad + k0 + (hs << 3);
  v8h lo = *(const v8h*)p;
  v8h hi = *(const v8h*)(p + 16);
  return __builtin_shufflevector(lo, hi, 0, 1, 2, 3, 4, 5, 6, 7,
                                 8, 9, 10, 11, 12, 13, 14, 15);
}

__device__ __forceinline__ unsigned pack2(float a, float b) {
  unsigned short ua = __builtin_bit_cast(unsigned short, (h16)a);
  unsigned short ub = __builtin_bit_cast(unsigned short, (h16)b);
  return (unsigned)ua | ((unsigned)ub << 16);
}

// ---------------------------------------------------------------------------
// Prep: repack weights TRANSPOSED (f16, [N][Kpad]) into ws, zero padded.
//   W1t: 16 x 32   @ 0       (K=25 used; N=12 used)
//   W2t: 16 x 320  @ 512     (K=c*25+ky*5+kx; group structure zero-padded)
//   W3t: 32 x 192  @ 5632    (K = c*16+py*4+px flatten; N=30 used)
//   W4t: 16 x 32   @ 11776   (K=30 used; N=10 used)
// ---------------------------------------------------------------------------
__global__ void prep_weights(const float* __restrict__ H1w,
                             const float* __restrict__ H2w,
                             const float* __restrict__ H3w,
                             const float* __restrict__ outw,
                             h16* __restrict__ ws) {
  int t = blockIdx.x * blockDim.x + threadIdx.x;
  if (t < 512) {                               // W1t
    int n = t >> 5, k = t & 31;
    ws[t] = (k < 25 && n < 12) ? (h16)H1w[n * 25 + k] : (h16)0.f;
  } else if (t < 5632) {                       // W2t
    int e = t - 512;
    int n = e / 320, k = e - n * 320;
    float v = 0.f;
    if (k < 300 && n < 12) {
      int c = k / 25, r = k - 25 * c;          // global in-channel, 5x5 idx
      int j = -1;
      if (n < 4)      { if (c < 8) j = c; }
      else if (n < 8) { if (c >= 4 && c < 12) j = c - 4; }
      else            { if (c < 4) j = c; else if (c >= 8) j = c - 4; }
      if (j >= 0) v = H2w[(n * 8 + j) * 25 + r];
    }
    ws[512 + e] = (h16)v;
  } else if (t < 11776) {                      // W3t
    int e = t - 5632;
    int n = e / 192, k = e - n * 192;
    ws[5632 + e] = (n < 30) ? (h16)H3w[k * 30 + n] : (h16)0.f;
  } else if (t < 12288) {                      // W4t
    int e = t - 11776;
    int n = e >> 5, k = e & 31;
    ws[11776 + e] = (k < 30 && n < 10) ? (h16)outw[k * 10 + n] : (h16)0.f;
  }
}

// ---------------------------------------------------------------------------
// Fused net, one wave = one 16-image tile, all MACs on v_wmma.
// ---------------------------------------------------------------------------
__global__ __launch_bounds__(WAVES * 32)
void lenet_fused(const float* __restrict__ x,
                 const float* __restrict__ H1b, const float* __restrict__ H2b,
                 const float* __restrict__ H3b, const float* __restrict__ outb,
                 const h16* __restrict__ W1, const h16* __restrict__ W2,
                 const h16* __restrict__ W3, const h16* __restrict__ W4,
                 float* __restrict__ out, int nimg) {
  // pre-padded activation storage (pad value -1 baked in)
  __shared__ __align__(16) h16 s_xs[WAVES][16 * 400];   // [img][20*20]
  __shared__ __align__(16) h16 s_a1[WAVES][16 * 1728];  // [img][12ch][12][12]
  __shared__ __align__(16) h16 s_a2[WAVES][16 * 192];   // [img][c*16+qy*4+qx]
  __shared__ __align__(16) h16 s_a3[WAVES][16 * 32];    // [img][n]
  __shared__ float s_bias[1024];  // [0,768): H1b  [768,960): H2b
                                  // [960,992): H3b  [992,1008): outb

  const int lane = threadIdx.x & 31;
  const int wv   = threadIdx.x >> 5;
  const int hs   = lane >> 4;      // wave-half select
  const int ncol = lane & 15;      // B/C/D column
  const int mrow = lane & 15;      // A row (image in tile)
  const v8f vzero = {};

  int img0 = (blockIdx.x * WAVES + wv) * 16;
  if (img0 + 16 > nimg) img0 = nimg - 16;

  // ---- fills: -1 borders + bias table + x interior ------------------------
  {
    unsigned ff = 0xBC00BC00u;                 // two f16 -1.0
    uint4 neg1 = make_uint4(ff, ff, ff, ff);
    uint4* p1 = (uint4*)&s_xs[wv][0];          // 800 uint4
    for (int i = lane; i < 800; i += 32) p1[i] = neg1;
    uint4* p2 = (uint4*)&s_a1[wv][0];          // 3456 uint4
    for (int i = lane; i < 3456; i += 32) p2[i] = neg1;

    for (int i = threadIdx.x; i < 1024; i += WAVES * 32) {
      float v = 0.f;
      if (i < 768)      v = H1b[i];
      else if (i < 960) v = H2b[i - 768];
      else if (i < 992) v = (i - 960 < 30) ? H3b[i - 960] : 0.f;
      else if (i < 1008) v = (i - 992 < 10) ? outb[i - 992] : 0.f;
      s_bias[i] = v;
    }

    const float4* xv = (const float4*)(x + (size_t)img0 * 256);
    for (int t = lane; t < 1024; t += 32) {    // 16 img * 64 float4
      int img = t >> 6, rem = t & 63;
      int y = rem >> 2, xq = rem & 3;
      float4 f = xv[t];
      int off = img * 400 + (y + 2) * 20 + 2 + xq * 4;   // even half index
      unsigned* dst = (unsigned*)&s_xs[wv][off];
      dst[0] = pack2(f.x, f.y);
      dst[1] = pack2(f.z, f.w);
    }
  }
  __syncthreads();

  // ---- H1: 64 positions, WMMA 16x16x32 (K=25 used) ------------------------
  {
    v16h b1 = load_bfrag(W1, 32, 0, 0, ncol, hs);
    int base1[16];                              // per-lane gather bases
#pragma unroll
    for (int i = 0; i < 16; ++i) {
      int k0 = (i & 7) + ((i >> 3) << 4);
      int k1 = k0 + 8;
      if (k0 > 24) k0 = 24;                     // rows >=25 are zero in W1
      if (k1 > 24) k1 = 24;
      int d0 = (k0 / 5) * 20 + (k0 % 5);
      int d1 = (k1 / 5) * 20 + (k1 % 5);
      base1[i] = mrow * 400 + d0 + hs * (d1 - d0);
    }
#pragma unroll
    for (int py = 0; py < 8; ++py) {
#pragma unroll
      for (int px = 0; px < 8; ++px) {
        const int poff = (2 * py) * 20 + 2 * px;   // folds into DS imm offset
        v16h a;
#pragma unroll
        for (int i = 0; i < 16; ++i) a[i] = s_xs[wv][base1[i] + poff];
        v8f acc = wmma_f16(a, b1, vzero);
        int p = py * 8 + px;
        float bias = s_bias[ncol * 64 + p];
        float tv[8];
#pragma unroll
        for (int r = 0; r < 8; ++r) tv[r] = tanh_fast(acc[r] + bias);
        if (ncol < 12) {
          int sb = ncol * 144 + (py + 2) * 12 + (px + 2);
#pragma unroll
          for (int r = 0; r < 8; ++r)
            s_a1[wv][(r + (hs << 3)) * 1728 + sb] = (h16)tv[r];
        }
      }
    }
  }
  __syncthreads();

  // ---- H2: K=320 in 10 chunks x 16 positions ------------------------------
  {
    v8f acc[16];
#pragma unroll
    for (int q = 0; q < 16; ++q) acc[q] = vzero;
#pragma unroll
    for (int kk = 0; kk < 10; ++kk) {
      v16h b = load_bfrag(W2, 320, 0, kk * 32, ncol, hs);
      int base2[16];
#pragma unroll
      for (int i = 0; i < 16; ++i) {
        int k0 = kk * 32 + (i & 7) + ((i >> 3) << 4);
        int k1 = k0 + 8;
        if (k0 >= 300) k0 = 0;                  // rows >=300 are zero in W2
        if (k1 >= 300) k1 = 0;
        int c0 = k0 / 25, r0 = k0 - 25 * c0;
        int c1 = k1 / 25, r1 = k1 - 25 * c1;
        int d0 = c0 * 144 + (r0 / 5) * 12 + (r0 % 5);
        int d1 = c1 * 144 + (r1 / 5) * 12 + (r1 % 5);
        base2[i] = mrow * 1728 + d0 + hs * (d1 - d0);
      }
#pragma unroll
      for (int q = 0; q < 16; ++q) {
        const int qoff = ((q >> 2) * 2) * 12 + (q & 3) * 2;  // DS imm offset
        v16h a;
#pragma unroll
        for (int i = 0; i < 16; ++i) a[i] = s_a1[wv][base2[i] + qoff];
        acc[q] = wmma_f16(a, b, acc[q]);
      }
    }
#pragma unroll
    for (int q = 0; q < 16; ++q) {
      float bias = s_bias[768 + ncol * 16 + q];
      float tv[8];
#pragma unroll
      for (int r = 0; r < 8; ++r) tv[r] = tanh_fast(acc[q][r] + bias);
      if (ncol < 12) {
#pragma unroll
        for (int r = 0; r < 8; ++r)
          s_a2[wv][(r + (hs << 3)) * 192 + ncol * 16 + q] = (h16)tv[r];
      }
    }
  }
  __syncthreads();

  // ---- H3: [16x192] @ [192x30] -> tanh -> s_a3 ----------------------------
  {
    v8f acc0 = vzero, acc1 = vzero;
    const int base3 = mrow * 192 + (hs << 3);   // 16B aligned
#pragma unroll
    for (int kc = 0; kc < 6; ++kc) {
      v8h lo = *(const v8h*)&s_a2[wv][base3 + kc * 32];
      v8h hi = *(const v8h*)&s_a2[wv][base3 + kc * 32 + 16];
      v16h a = __builtin_shufflevector(lo, hi, 0, 1, 2, 3, 4, 5, 6, 7,
                                       8, 9, 10, 11, 12, 13, 14, 15);
      acc0 = wmma_f16(a, load_bfrag(W3, 192, 0,  kc * 32, ncol, hs), acc0);
      acc1 = wmma_f16(a, load_bfrag(W3, 192, 16, kc * 32, ncol, hs), acc1);
    }
    float bias0 = s_bias[960 + ncol];
#pragma unroll
    for (int r = 0; r < 8; ++r)
      s_a3[wv][(r + (hs << 3)) * 32 + ncol] = (h16)tanh_fast(acc0[r] + bias0);
    int n2 = 16 + ncol;
    float bias1 = s_bias[960 + n2];
#pragma unroll
    for (int r = 0; r < 8; ++r) {
      float tv = tanh_fast(acc1[r] + bias1);
      h16 v = (n2 < 30) ? (h16)tv : (h16)0.f;   // zero K rows 30..31 for W4
      s_a3[wv][(r + (hs << 3)) * 32 + n2] = v;
    }
  }
  __syncthreads();

  // ---- output layer: [16x30] @ [30x10] -> global --------------------------
  {
    const int base4 = mrow * 32 + (hs << 3);    // 16B aligned
    v8h lo = *(const v8h*)&s_a3[wv][base4];
    v8h hi = *(const v8h*)&s_a3[wv][base4 + 16];
    v16h a = __builtin_shufflevector(lo, hi, 0, 1, 2, 3, 4, 5, 6, 7,
                                     8, 9, 10, 11, 12, 13, 14, 15);
    v16h b = load_bfrag(W4, 32, 0, 0, ncol, hs);
    v8f acc = wmma_f16(a, b, vzero);
    float bias = s_bias[992 + ncol];
    float tv[8];
#pragma unroll
    for (int r = 0; r < 8; ++r) tv[r] = tanh_fast(acc[r] + bias);
    if (ncol < 10) {
#pragma unroll
      for (int r = 0; r < 8; ++r) {
        int m = r + (hs << 3);
        out[(size_t)(img0 + m) * 10 + ncol] = tv[r];
      }
    }
  }
}

extern "C" void kernel_launch(void* const* d_in, const int* in_sizes, int n_in,
                              void* d_out, int out_size, void* d_ws, size_t ws_size,
                              hipStream_t stream) {
  const float* x    = (const float*)d_in[0];
  const float* H1w  = (const float*)d_in[1];
  const float* H1b  = (const float*)d_in[2];
  const float* H2w  = (const float*)d_in[3];
  const float* H2b  = (const float*)d_in[4];
  const float* H3w  = (const float*)d_in[5];
  const float* H3b  = (const float*)d_in[6];
  const float* outw = (const float*)d_in[7];
  const float* outb = (const float*)d_in[8];

  h16* ws = (h16*)d_ws;
  const h16* W1 = ws;
  const h16* W2 = ws + 512;
  const h16* W3 = ws + 5632;
  const h16* W4 = ws + 11776;

  prep_weights<<<48, 256, 0, stream>>>(H1w, H2w, H3w, outw, ws);

  int nimg = in_sizes[0] / 256;                        // 65536
  int grid = (nimg + WAVES * 16 - 1) / (WAVES * 16);   // 2048
  lenet_fused<<<grid, WAVES * 32, 0, stream>>>(
      x, H1b, H2b, H3b, outb, W1, W2, W3, W4, (float*)d_out, nimg);
}
